// SparseMoEBlock_40785009442950
// MI455X (gfx1250) — compile-verified, matrix operand order checked by reading, not run on hardware
//
#include <hip/hip_runtime.h>
#include <math.h>

#define B_    2
#define S_    2048
#define H_    1024
#define HFF_  4096
#define E_    8
#define NTOK  (B_ * S_)       // 4096 tokens
#define WAVES 8
#define BLOCK 256
#define HSTRIDE (HFF_ + 8)    // padded LDS row stride (8208 B -> 4-dword bank rotation/row)

typedef __attribute__((ext_vector_type(8)))  __bf16 v8bf;
typedef __attribute__((ext_vector_type(16))) __bf16 v16bf;
typedef __attribute__((ext_vector_type(8)))  float  v8f;

// Build a 16-element bf16 WMMA fragment for one lane.
// Per CDNA5 ISA 16-bit A/B layout: a lane holds K in two contiguous 8-element
// (16B) runs: [kbase, kbase+8) and [kbase+16, kbase+24).
__device__ __forceinline__ v16bf make_frag(const __bf16* p) {
    v8bf lo = *(const v8bf*)(p);
    v8bf hi = *(const v8bf*)(p + 16);
    return __builtin_shufflevector(lo, hi,
        0, 1, 2, 3, 4, 5, 6, 7, 8, 9, 10, 11, 12, 13, 14, 15);
}

__device__ __forceinline__ float gelu_tanh(float v) {
    return 0.5f * v * (1.0f + tanhf(0.7978845608028654f * (v + 0.044715f * v * v * v)));
}

// 4-N-tile GEMM K-sweep: one shared A fragment feeds 4 WMMAs; manual ping-pong
// (unroll-by-2) double buffering so buffer rotation is register renaming, not
// v_mov copies. B row stride for tile t is 16*KTOT.
template <int KTOT>
__device__ __forceinline__ void gemm_4tile(const __bf16* __restrict__ abase,
                                           const __bf16* __restrict__ bbase,
                                           v8f acc[4]) {
    v16bf aP = make_frag(abase);
    v16bf bP[4];
#pragma unroll
    for (int t = 0; t < 4; ++t) bP[t] = make_frag(bbase + (size_t)t * 16 * KTOT);

    for (int ks = 0; ks < KTOT - 64; ks += 64) {
        v16bf aQ = make_frag(abase + ks + 32);
        v16bf bQ[4];
#pragma unroll
        for (int t = 0; t < 4; ++t)
            bQ[t] = make_frag(bbase + (size_t)t * 16 * KTOT + ks + 32);
#pragma unroll
        for (int t = 0; t < 4; ++t)
            acc[t] = __builtin_amdgcn_wmma_f32_16x16x32_bf16(
                false, aP, false, bP[t], (short)0, acc[t], false, false);

        aP = make_frag(abase + ks + 64);
#pragma unroll
        for (int t = 0; t < 4; ++t)
            bP[t] = make_frag(bbase + (size_t)t * 16 * KTOT + ks + 64);
#pragma unroll
        for (int t = 0; t < 4; ++t)
            acc[t] = __builtin_amdgcn_wmma_f32_16x16x32_bf16(
                false, aQ, false, bQ[t], (short)0, acc[t], false, false);
    }
    // epilogue: P holds step KTOT-64; load and finish step KTOT-32
    {
        v16bf aQ = make_frag(abase + KTOT - 32);
        v16bf bQ[4];
#pragma unroll
        for (int t = 0; t < 4; ++t)
            bQ[t] = make_frag(bbase + (size_t)t * 16 * KTOT + KTOT - 32);
#pragma unroll
        for (int t = 0; t < 4; ++t)
            acc[t] = __builtin_amdgcn_wmma_f32_16x16x32_bf16(
                false, aP, false, bP[t], (short)0, acc[t], false, false);
#pragma unroll
        for (int t = 0; t < 4; ++t)
            acc[t] = __builtin_amdgcn_wmma_f32_16x16x32_bf16(
                false, aQ, false, bQ[t], (short)0, acc[t], false, false);
    }
}

// ---------------------------------------------------------------------------
__global__ void init_ws(float* __restrict__ out, int* __restrict__ counts,
                        float* __restrict__ gatelist) {
    size_t i = (size_t)blockIdx.x * blockDim.x + threadIdx.x;
    size_t stride = (size_t)gridDim.x * blockDim.x;
    for (size_t j = i; j < (size_t)NTOK * H_; j += stride) out[j] = 0.0f;
    for (size_t j = i; j < (size_t)E_ * NTOK; j += stride) gatelist[j] = 0.0f;
    if (i < E_) counts[i] = 0;
}

// ---------------------------------------------------------------------------
__global__ void f32_to_bf16(const float* __restrict__ in, __bf16* __restrict__ out,
                            int n) {
    int i = blockIdx.x * blockDim.x + threadIdx.x;
    if (i < n) out[i] = (__bf16)in[i];
}

// ---------------------------------------------------------------------------
// Token routing: gate[e] = max_k probs[k] * (idx[k]==e); duplicates collapse to
// a single entry with max prob -> matches the reference max-over-k exactly.
__global__ void route(const float* __restrict__ probs, const int* __restrict__ idx,
                      int* __restrict__ counts, int* __restrict__ toklist,
                      float* __restrict__ gatelist) {
    int t = blockIdx.x * blockDim.x + threadIdx.x;
    if (t >= NTOK) return;
    float p0 = probs[t * 2 + 0];
    float p1 = probs[t * 2 + 1];
    int   e0 = idx[t * 2 + 0];
    int   e1 = idx[t * 2 + 1];
    if (e0 == e1) {
        int s = atomicAdd(&counts[e0], 1);
        toklist[e0 * NTOK + s]  = t;
        gatelist[e0 * NTOK + s] = fmaxf(p0, p1);
    } else {
        int s0 = atomicAdd(&counts[e0], 1);
        toklist[e0 * NTOK + s0]  = t;
        gatelist[e0 * NTOK + s0] = p0;
        int s1 = atomicAdd(&counts[e1], 1);
        toklist[e1 * NTOK + s1]  = t;
        gatelist[e1 * NTOK + s1] = p1;
    }
}

// ---------------------------------------------------------------------------
// Fused expert FFN: per (expert, 16-token tile) workgroup, 8 wave32s.
// Phase 1: full gelu(x @ W1^T) tile (16 x HFF) -> LDS (128 KB, bank-padded).
// Phase 2: y = h @ W2^T, batches of 4 N-tiles per wave, then gated scatter.
__global__ __launch_bounds__(BLOCK) void moe_ffn(
    const __bf16* __restrict__ xb,   const __bf16* __restrict__ w1b,
    const __bf16* __restrict__ w2b,  const int* __restrict__ counts,
    const int* __restrict__ toklist, const float* __restrict__ gatelist,
    float* __restrict__ out) {

    const int e  = blockIdx.y;
    const int mt = blockIdx.x;
    const int count = counts[e];
    if (mt * 16 >= count) return;          // block-uniform early exit

    __shared__ int    toks[16];
    __shared__ float  gates[16];
    __shared__ __bf16 hbuf[16 * HSTRIDE];  // gelu(h) tile

    const int tid = threadIdx.x;
    if (tid < 16) {
        int slot  = mt * 16 + tid;
        int valid = slot < count;
        toks[tid]  = toklist[e * NTOK + (valid ? slot : 0)];
        gates[tid] = valid ? gatelist[e * NTOK + slot] : 0.0f;
    }
    __syncthreads();

    const int wave = tid >> 5;
    const int lane = tid & 31;
    const int half = lane >> 4;            // 0: lanes 0-15, 1: lanes 16-31
    const int l16  = lane & 15;
    const int kofs = half * 8;             // K sub-offset per ISA layout

    const __bf16* w1e  = w1b + (size_t)e * HFF_ * H_;
    const __bf16* w2e  = w2b + (size_t)e * H_ * HFF_;
    const __bf16* xrow = xb + (size_t)toks[l16] * H_ + kofs;  // A base, GEMM1

    // ---- Phase 1: h = gelu(x @ W1^T). 256 N-tiles -> 8 batches of 4/wave ----
    for (int nb = 0; nb < (HFF_ / 16) / (WAVES * 4); ++nb) {
        const int tile0 = nb * (WAVES * 4) + wave * 4;   // 4 consecutive tiles
        const __bf16* br = w1e + ((size_t)tile0 * 16 + l16) * H_ + kofs;

        v8f acc[4] = {};
        gemm_4tile<H_>(xrow, br, acc);

#pragma unroll
        for (int t = 0; t < 4; ++t) {
            const int fcol = (tile0 + t) * 16 + l16;
#pragma unroll
            for (int r = 0; r < 8; ++r) {
                int row = r + half * 8;
                hbuf[row * HSTRIDE + fcol] = (__bf16)gelu_tanh(acc[t][r]);
            }
        }
    }
    __syncthreads();

    // ---- Phase 2: y = h @ W2^T. 64 N-tiles -> 2 batches of 4/wave ----
    const __bf16* arow = hbuf + l16 * HSTRIDE + kofs;
    for (int nb = 0; nb < (H_ / 16) / (WAVES * 4); ++nb) {
        const int tile0 = nb * (WAVES * 4) + wave * 4;
        const __bf16* br = w2e + ((size_t)tile0 * 16 + l16) * HFF_ + kofs;

        v8f acc[4] = {};
        gemm_4tile<HFF_>(arow, br, acc);

        // gated scatter-accumulate (a token may appear in 2 expert lists)
#pragma unroll
        for (int t = 0; t < 4; ++t) {
            const int j = (tile0 + t) * 16 + l16;
#pragma unroll
            for (int r = 0; r < 8; ++r) {
                int row = r + half * 8;
                atomicAdd(&out[(size_t)toks[row] * H_ + j], gates[row] * acc[t][r]);
            }
        }
    }
}

// ---------------------------------------------------------------------------
extern "C" void kernel_launch(void* const* d_in, const int* in_sizes, int n_in,
                              void* d_out, int out_size, void* d_ws, size_t ws_size,
                              hipStream_t stream) {
    const float* x     = (const float*)d_in[0];
    const float* probs = (const float*)d_in[1];
    const int*   idx   = (const int*)d_in[2];
    const float* W1    = (const float*)d_in[3];
    const float* W2    = (const float*)d_in[4];
    float*       out   = (float*)d_out;

    // Workspace layout (all 16B aligned): xb | w1b | w2b | counts | toklist | gatelist
    char* ws = (char*)d_ws;
    __bf16* xb      = (__bf16*)ws;  ws += (size_t)NTOK * H_ * sizeof(__bf16);
    __bf16* w1b     = (__bf16*)ws;  ws += (size_t)E_ * HFF_ * H_ * sizeof(__bf16);
    __bf16* w2b     = (__bf16*)ws;  ws += (size_t)E_ * H_ * HFF_ * sizeof(__bf16);
    int*    counts  = (int*)ws;     ws += 64;
    int*    toklist = (int*)ws;     ws += (size_t)E_ * NTOK * sizeof(int);
    float*  gatelist= (float*)ws;

    init_ws<<<2048, 256, 0, stream>>>(out, counts, gatelist);

    const int nx = NTOK * H_;          // 4.19M
    const int nw = E_ * HFF_ * H_;     // 33.6M
    f32_to_bf16<<<(nx + 255) / 256, 256, 0, stream>>>(x,  xb,  nx);
    f32_to_bf16<<<(nw + 255) / 256, 256, 0, stream>>>(W1, w1b, nw);
    f32_to_bf16<<<(nw + 255) / 256, 256, 0, stream>>>(W2, w2b, nw);

    route<<<(NTOK + 255) / 256, 256, 0, stream>>>(probs, idx, counts, toklist, gatelist);

    dim3 grid(NTOK / 16, E_);          // 256 M-tiles (early-exit past count) x 8 experts
    moe_ffn<<<grid, BLOCK, 0, stream>>>(xb, w1b, w2b, counts, toklist, gatelist, out);
}
